// FSGPTMoEBlock_57818849739129
// MI455X (gfx1250) — matrix operand order repaired and checked by reading, not compiled, not run on hardware
//
#include <hip/hip_runtime.h>
#include <hip/hip_bf16.h>
#include <math.h>

// ---------------------------------------------------------------------------
// FSGPTMoEBlock for MI455X (gfx1250): WMMA f32_16x16x32_f16 everywhere,
// software-pipelined async global->LDS staging, b128 fragment loads,
// V stored transposed so attention PV operands are contiguous.
// ---------------------------------------------------------------------------

typedef __attribute__((ext_vector_type(16))) _Float16 v16h;
typedef __attribute__((ext_vector_type(8)))  float    v8f;

#define TT 2048   // B*S tokens
#define DD 1024   // model dim
#define FF 4096   // ffn dim
#define EE 16     // experts
#define CAP 256   // expert capacity = 2*ceil(T/E)
#define SS 1024   // seq len
#define HH 16     // heads
#define HD 64     // head dim
#define BB 2

#if defined(__has_builtin)
#if __has_builtin(__builtin_amdgcn_global_load_async_to_lds_b128)
#define ASYNC_LDS 1
#endif
#endif

#ifdef ASYNC_LDS
typedef int v4i_vs __attribute__((__vector_size__(16)));
typedef __attribute__((address_space(1))) v4i_vs* as1_v4i;
typedef __attribute__((address_space(3))) v4i_vs* as3_v4i;
#endif

__device__ __forceinline__ v8f wmma32(v16h a, v16h b, v8f c) {
  return __builtin_amdgcn_wmma_f32_16x16x32_f16(false, a, false, b, (short)0, c,
                                                false, false);
}

// Two 16B loads -> one v16h. off = element offset of the second 16B chunk.
__device__ __forceinline__ v16h ld2x16(const _Float16* p, int off) {
  union { v16h v; int4 q[2]; } u;
  u.q[0] = *(const int4*)p;
  u.q[1] = *(const int4*)(p + off);
  return u.v;
}

// A fragment (16x32 f16, row-major, leading dim ld):
// lane<16: row=lane, K=0..15; lane>=16: row=lane-16, K=16..31
__device__ __forceinline__ v16h frag_a(const _Float16* base, int ld) {
  const int lane = threadIdx.x & 31;
  return ld2x16(base + (lane & 15) * ld + ((lane >> 4) << 4), 8);
}

// B fragment from TRANSPOSED [N][K] tile (leading dim ld):
// lane<16: col=lane, K=0..15; lane>=16: col=lane-16, K=16..31 (contiguous!)
__device__ __forceinline__ v16h frag_bt(const _Float16* baseT, int ld) {
  const int lane = threadIdx.x & 31;
  return ld2x16(baseT + (lane & 15) * ld + ((lane >> 4) << 4), 8);
}

// 16B global -> LDS stage (async copy when the toolchain exposes it)
__device__ __forceinline__ void stage16(const _Float16* g, _Float16* l) {
#ifdef ASYNC_LDS
  __builtin_amdgcn_global_load_async_to_lds_b128((as1_v4i)g, (as3_v4i)l, 0, 0);
#else
  *(int4*)l = *(const int4*)g;
#endif
}

__device__ __forceinline__ void wait_async0() {
#ifdef ASYNC_LDS
#if __has_builtin(__builtin_amdgcn_s_wait_asynccnt)
  __builtin_amdgcn_s_wait_asynccnt(0);
#else
  asm volatile("s_wait_asynccnt 0x0" ::: "memory");
#endif
#endif
}

// allow the newest 4 async copies (next tile) to stay in flight
__device__ __forceinline__ void wait_async4() {
#ifdef ASYNC_LDS
#if __has_builtin(__builtin_amdgcn_s_wait_asynccnt)
  __builtin_amdgcn_s_wait_asynccnt(4);
#else
  asm volatile("s_wait_asynccnt 0x4" ::: "memory");
#endif
#endif
}

// ---------------------------------------------------------------------------
// LayerNorm: f32 row -> f16 normalized row
// ---------------------------------------------------------------------------
__global__ void __launch_bounds__(256)
ln_kernel(const float* __restrict__ x, const float* __restrict__ w,
          const float* __restrict__ b, _Float16* __restrict__ out) {
  __shared__ float sh[256], sh2[256];
  const long t = blockIdx.x;
  const float* row = x + t * DD;
  float s = 0.f, s2 = 0.f;
  for (int i = threadIdx.x; i < DD; i += 256) {
    float v = row[i]; s += v; s2 += v * v;
  }
  sh[threadIdx.x] = s; sh2[threadIdx.x] = s2;
  __syncthreads();
  for (int off = 128; off > 0; off >>= 1) {
    if ((int)threadIdx.x < off) {
      sh[threadIdx.x] += sh[threadIdx.x + off];
      sh2[threadIdx.x] += sh2[threadIdx.x + off];
    }
    __syncthreads();
  }
  const float mu = sh[0] * (1.f / DD);
  const float var = sh2[0] * (1.f / DD) - mu * mu;
  const float inv = rsqrtf(var + 1e-5f);
  for (int i = threadIdx.x; i < DD; i += 256)
    out[t * DD + i] = (_Float16)((row[i] - mu) * inv * w[i] + b[i]);
}

// ---------------------------------------------------------------------------
// Generic WMMA GEMM: C[z] = act(A[z](f16) * B[z](f32->f16) + bias[z]) * scale
// Block tile 256(M) x 64(N); 8 waves, each wave owns 32x64 (8 accumulators)
// -> 8 v_wmma per K-step per wave. Software pipeline: A tiles triple-buffered
// via async global->LDS (tile i+1 in flight during compute of tile i,
// s_wait_asynccnt 4); B tiles double-buffered through registers; one barrier
// per K-step. VT: store result as transposed V layout [b][h][d][s].
// ---------------------------------------------------------------------------
template <bool GELU, bool VT>
__global__ void __launch_bounds__(256)
gemm_kernel(const _Float16* __restrict__ A, long strideA, int lda,
            const float* __restrict__ B, long strideB, int ldb,
            const float* __restrict__ bias, long strideBias, float scale,
            _Float16* __restrict__ Ch, float* __restrict__ Cf, long strideC,
            int ldc, const float* __restrict__ resid, int K) {
  __shared__ __align__(16) _Float16 As[3][256][40];  // triple-buffered A
  __shared__ __align__(16) _Float16 Bst[2][64][40];  // double-buffered B^T
  const int tid = threadIdx.x;
  const int z = blockIdx.z;
  A += (long)z * strideA;
  B += (long)z * strideB;
  if (bias) bias += (long)z * strideBias;
  const int m0 = blockIdx.y * 256;
  const int n0 = blockIdx.x * 64;
  const int w = tid >> 5;
  const int wm = w * 32;                        // wave rows: wm..wm+31
  const int bk = tid >> 3, bn = (tid & 7) * 8;  // B-stage coords
  v8f acc[2][4] = {};
  float4 f0, f1;
  auto loadB = [&](int kk) {
    const float* bsrc = B + (long)(kk + bk) * ldb + n0 + bn;
    f0 = *(const float4*)bsrc;
    f1 = *(const float4*)(bsrc + 4);
  };
  auto issueA = [&](int kk, int buf) {
    const _Float16* asrc = A + (long)(m0 + tid) * lda + kk;
#pragma unroll
    for (int c = 0; c < 4; ++c) stage16(asrc + c * 8, &As[buf][tid][c * 8]);
  };
  issueA(0, 0);
  loadB(0);
  const int steps = K >> 5;
  for (int it = 0; it < steps; ++it) {
    const int cur = it % 3;
    const int bb = it & 1;
    // commit staged B registers (f32->f16, transposed) into Bst[bb]
    Bst[bb][bn + 0][bk] = (_Float16)f0.x; Bst[bb][bn + 1][bk] = (_Float16)f0.y;
    Bst[bb][bn + 2][bk] = (_Float16)f0.z; Bst[bb][bn + 3][bk] = (_Float16)f0.w;
    Bst[bb][bn + 4][bk] = (_Float16)f1.x; Bst[bb][bn + 5][bk] = (_Float16)f1.y;
    Bst[bb][bn + 6][bk] = (_Float16)f1.z; Bst[bb][bn + 7][bk] = (_Float16)f1.w;
    if (it + 1 < steps) {           // prefetch tile it+1 while computing it
      issueA(it * 32 + 32, (it + 1) % 3);
      loadB(it * 32 + 32);
      wait_async4();                // current tile's copies complete
    } else {
      wait_async0();
    }
    __syncthreads();
    const v16h a0 = frag_a(&As[cur][wm][0], 40);
    const v16h a1 = frag_a(&As[cur][wm + 16][0], 40);
#pragma unroll
    for (int nt = 0; nt < 4; ++nt) {
      const v16h b = frag_bt(&Bst[bb][nt * 16][0], 40);
      acc[0][nt] = wmma32(a0, b, acc[0][nt]);
      acc[1][nt] = wmma32(a1, b, acc[1][nt]);
    }
  }
  // epilogue (C layout: lane<16 -> col=lane rows 0..7; lane>=16 -> rows 8..15)
  const int lane = tid & 31;
  const int col = lane & 15;
  const int rb = (lane >> 4) << 3;
#pragma unroll
  for (int nt = 0; nt < 4; ++nt) {
    const int cg = n0 + nt * 16 + col;
    const float bval = bias ? bias[cg] : 0.f;
#pragma unroll
    for (int sub = 0; sub < 2; ++sub) {
      v8f a = acc[sub][nt];
#pragma unroll
      for (int r = 0; r < 8; ++r) {
        const int rg = m0 + wm + sub * 16 + rb + r;
        float val = a[r] + bval;
        if (GELU) val = 0.5f * val * (1.f + erff(val * 0.70710678f));
        val *= scale;
        if (resid) val += resid[(long)rg * ldc + cg];
        if (VT) {
          // V^T layout: [b][h][d][s]; cg = h*64+d, rg = b*1024+s
          const long vidx =
              (((long)(rg >> 10) * HH + (cg >> 6)) * HD + (cg & 63)) * SS +
              (rg & (SS - 1));
          Ch[vidx] = (_Float16)val;
        } else {
          const long idx = (long)z * strideC + (long)rg * ldc + cg;
          if (Ch) Ch[idx] = (_Float16)val;
          else    Cf[idx] = val;
        }
      }
    }
  }
}

// ---------------------------------------------------------------------------
// Flash attention: one wave per 16-row q block; 32-key chunks; online softmax.
// q already scaled by 1/sqrt(HD). q/k/ctx are f16 [T, D] (head-packed D);
// v is f16 transposed [b][h][d][s] so PV B-fragments are contiguous.
// ---------------------------------------------------------------------------
__global__ void __launch_bounds__(32)
attn_kernel(const _Float16* __restrict__ q, const _Float16* __restrict__ k,
            const _Float16* __restrict__ vt, _Float16* __restrict__ ctx) {
  const int q0 = blockIdx.x * 16;
  const int b = blockIdx.y >> 4, h = blockIdx.y & 15;
  const long rowbase = (long)b * SS;
  const int lane = threadIdx.x & 31;
  const int col = lane & 15;
  const int rb = (lane >> 4) << 3;
  const int kh8 = (lane >> 4) << 3;
  const int kh16 = (lane >> 4) << 4;
  __shared__ __align__(16) _Float16 pl[16][40];

  // q fragments: rows q0..q0+15, head dims 0..31 and 32..63
  const _Float16* qp = q + (rowbase + q0 + col) * (long)DD + h * HD + kh16;
  const v16h aq0 = ld2x16(qp, 8);
  const v16h aq1 = ld2x16(qp + 32, 8);
  const _Float16* vtb = vt + ((long)(b * HH + h) * HD) * SS;

  float mrow[8], lrow[8];
#pragma unroll
  for (int r = 0; r < 8; ++r) { mrow[r] = -1e30f; lrow[r] = 0.f; }
  v8f cacc[4] = {};

  for (int kc = 0; kc <= q0 + 15; kc += 32) {
    const bool act1 = (kc + 16) <= (q0 + 15);
    v8f s0 = {}; v8f s1 = {};
    {  // keys kc..kc+15 : scores += q * k^T (k rows are contiguous per lane)
      const _Float16* kp = k + (rowbase + kc + col) * (long)DD + h * HD + kh16;
      s0 = wmma32(aq0, ld2x16(kp, 8), s0);
      s0 = wmma32(aq1, ld2x16(kp + 32, 8), s0);
    }
    if (act1) {  // keys kc+16..kc+31
      const _Float16* kp =
          k + (rowbase + kc + 16 + col) * (long)DD + h * HD + kh16;
      s1 = wmma32(aq0, ld2x16(kp, 8), s1);
      s1 = wmma32(aq1, ld2x16(kp + 32, 8), s1);
    }
    // causal mask + online softmax (row r lives across a 16-lane half)
#pragma unroll
    for (int r = 0; r < 8; ++r) {
      const int qr = q0 + rb + r;
      if (kc + col > qr) s0[r] = -1e30f;
      if (!act1 || (kc + 16 + col > qr)) s1[r] = -1e30f;
      float tm = fmaxf(s0[r], s1[r]);
#pragma unroll
      for (int off = 8; off > 0; off >>= 1)
        tm = fmaxf(tm, __shfl_xor(tm, off, 16));
      const float nm = fmaxf(mrow[r], tm);
      const float alpha = __expf(mrow[r] - nm);
      const float p0 = __expf(s0[r] - nm);
      const float p1 = __expf(s1[r] - nm);
      float ps = p0 + p1;
#pragma unroll
      for (int off = 8; off > 0; off >>= 1) ps += __shfl_xor(ps, off, 16);
      lrow[r] = lrow[r] * alpha + ps;
      mrow[r] = nm;
#pragma unroll
      for (int nt = 0; nt < 4; ++nt) cacc[nt][r] *= alpha;
      s0[r] = p0; s1[r] = p1;
    }
    // P -> LDS (C layout -> row-major 16x32), then re-fragment as A operand
#pragma unroll
    for (int r = 0; r < 8; ++r) {
      pl[rb + r][col] = (_Float16)s0[r];
      pl[rb + r][16 + col] = (_Float16)s1[r];
    }
    __syncthreads();
    const v16h ap = frag_a(&pl[0][0], 40);
    // base clamp: the only group that can start past S-16 is fully masked
    int kb = kc + kh16;
    if (kb > SS - 16) kb = SS - 16;
#pragma unroll
    for (int nt = 0; nt < 4; ++nt) {
      const v16h bv = ld2x16(vtb + (long)(nt * 16 + col) * SS + kb, 8);
      cacc[nt] = wmma32(ap, bv, cacc[nt]);
    }
    __syncthreads();
  }
  // normalize + store
#pragma unroll
  for (int r = 0; r < 8; ++r) {
    const float inv = 1.f / lrow[r];
    const long base = (rowbase + q0 + rb + r) * (long)DD + h * HD;
#pragma unroll
    for (int nt = 0; nt < 4; ++nt)
      ctx[base + nt * 16 + col] = (_Float16)(cacc[nt][r] * inv);
  }
}

// ---------------------------------------------------------------------------
// Router logits: [T,D](f16) @ [D,16](f32) -> [T,16](f32), one wave / 16 tokens
// ---------------------------------------------------------------------------
__global__ void __launch_bounds__(32)
router_kernel(const _Float16* __restrict__ x2, const float* __restrict__ rw,
              float* __restrict__ logits) {
  const int t0 = blockIdx.x * 16;
  const int lane = threadIdx.x & 31;
  const int col = lane & 15;
  const int kh8 = (lane >> 4) << 3;
  const int kh16 = (lane >> 4) << 4;
  v8f acc = {};
  for (int k0 = 0; k0 < DD; k0 += 32) {
    const v16h a = ld2x16(x2 + (long)(t0 + col) * DD + k0 + kh16, 8);
    v16h b;
#pragma unroll
    for (int i = 0; i < 16; ++i)
      b[i] = (_Float16)rw[(long)(k0 + kh16 + i) * EE + col];
    acc = wmma32(a, b, acc);
  }
  const int rb = (lane >> 4) << 3;
#pragma unroll
  for (int r = 0; r < 8; ++r) logits[(long)(t0 + rb + r) * EE + col] = acc[r];
}

// ---------------------------------------------------------------------------
// Top-2 + capacity routing (single block; sequential token-order cumsum)
// ---------------------------------------------------------------------------
__global__ void __launch_bounds__(256)
route_kernel(const float* __restrict__ logits, int* __restrict__ e1,
             int* __restrict__ e2, int* __restrict__ s1, int* __restrict__ s2,
             float* __restrict__ g1, float* __restrict__ g2) {
  __shared__ int tot1[EE];
  for (int t = threadIdx.x; t < TT; t += 256) {
    const float* l = logits + (long)t * EE;
    float mx = -3.4e38f; int a1 = 0;
#pragma unroll
    for (int e = 0; e < EE; ++e) if (l[e] > mx) { mx = l[e]; a1 = e; }
    float mx2 = -3.4e38f; int a2 = 0;
#pragma unroll
    for (int e = 0; e < EE; ++e)
      if (e != a1 && l[e] > mx2) { mx2 = l[e]; a2 = e; }
    float den = 0.f;
#pragma unroll
    for (int e = 0; e < EE; ++e) den += __expf(l[e] - mx);
    e1[t] = a1; e2[t] = a2;
    g1[t] = 1.f / den;               // prob of top1
    g2[t] = __expf(mx2 - mx) / den;  // prob of top2
  }
  if ((int)threadIdx.x < EE) tot1[threadIdx.x] = 0;
  __syncthreads();
  for (int t = threadIdx.x; t < TT; t += 256) atomicAdd(&tot1[e1[t]], 1);
  __syncthreads();
  if (threadIdx.x == 0) {
    int c1[EE] = {}, c2[EE] = {};
    for (int t = 0; t < TT; ++t) {
      const int a1 = e1[t], a2 = e2[t];
      const int pos1 = c1[a1]++;
      const int pos2 = c2[a2]++ + tot1[a2];  // 2nd choice queues after all top1
      const bool k1 = pos1 < CAP, k2 = pos2 < CAP;
      float p1 = k1 ? g1[t] : 0.f;
      float p2 = k2 ? g2[t] : 0.f;
      const float den = fmaxf(p1 + p2, 1.1920929e-07f);
      g1[t] = p1 / den; g2[t] = p2 / den;
      s1[t] = k1 ? pos1 : -1;
      s2[t] = k2 ? pos2 : -1;
    }
  }
}

__global__ void __launch_bounds__(256)
zero_f16_kernel(_Float16* __restrict__ p, long n) {
  for (long i = (long)blockIdx.x * 256 + threadIdx.x; i < n;
       i += (long)gridDim.x * 256)
    p[i] = (_Float16)0.f;
}

__global__ void __launch_bounds__(256)
gather_kernel(const _Float16* __restrict__ x2, const int* __restrict__ e1,
              const int* __restrict__ s1, const int* __restrict__ e2,
              const int* __restrict__ s2, _Float16* __restrict__ xin) {
  const long t = blockIdx.x;
  const _Float16* src = x2 + t * DD;
  int sl = s1[t];
  if (sl >= 0) {
    _Float16* d = xin + ((long)e1[t] * CAP + sl) * DD;
    for (int i = threadIdx.x; i < DD; i += 256) d[i] = src[i];
  }
  sl = s2[t];
  if (sl >= 0) {
    _Float16* d = xin + ((long)e2[t] * CAP + sl) * DD;
    for (int i = threadIdx.x; i < DD; i += 256) d[i] = src[i];
  }
}

__global__ void __launch_bounds__(256)
combine_kernel(const float* __restrict__ hidden, const float* __restrict__ eout,
               const int* __restrict__ e1, const int* __restrict__ s1,
               const int* __restrict__ e2, const int* __restrict__ s2,
               const float* __restrict__ g1, const float* __restrict__ g2,
               float* __restrict__ out) {
  const long t = blockIdx.x;
  const float w1 = g1[t], w2 = g2[t];
  const int p1 = s1[t], p2 = s2[t];
  const float* r1 = (p1 >= 0) ? eout + ((long)e1[t] * CAP + p1) * DD : nullptr;
  const float* r2 = (p2 >= 0) ? eout + ((long)e2[t] * CAP + p2) * DD : nullptr;
  for (int i = threadIdx.x; i < DD; i += 256) {
    float vv = hidden[t * DD + i];
    if (r1) vv += w1 * r1[i];
    if (r2) vv += w2 * r2[i];
    out[t * DD + i] = vv;
  }
}

// ---------------------------------------------------------------------------
extern "C" void kernel_launch(void* const* d_in, const int* in_sizes, int n_in,
                              void* d_out, int out_size, void* d_ws,
                              size_t ws_size, hipStream_t stream) {
  (void)in_sizes; (void)n_in; (void)out_size; (void)ws_size;
  const float* hs    = (const float*)d_in[0];
  const float* ln1w  = (const float*)d_in[1];
  const float* ln1b  = (const float*)d_in[2];
  const float* q_w   = (const float*)d_in[3];
  const float* q_b   = (const float*)d_in[4];
  const float* k_w   = (const float*)d_in[5];
  const float* k_b   = (const float*)d_in[6];
  const float* v_w   = (const float*)d_in[7];
  const float* v_b   = (const float*)d_in[8];
  const float* o_w   = (const float*)d_in[9];
  const float* o_b   = (const float*)d_in[10];
  const float* ln2w  = (const float*)d_in[11];
  const float* ln2b  = (const float*)d_in[12];
  const float* rw    = (const float*)d_in[13];
  const float* fc1w  = (const float*)d_in[14];
  const float* fc1b  = (const float*)d_in[15];
  const float* fc2w  = (const float*)d_in[16];
  const float* fc2b  = (const float*)d_in[17];
  float* out = (float*)d_out;

  // bump allocator over workspace (all chunks 256B aligned)
  char* wp = (char*)d_ws;
  auto alloc = [&](size_t bytes) {
    char* r = wp;
    wp += (bytes + 255) & ~(size_t)255;
    return (void*)r;
  };
  _Float16* x1   = (_Float16*)alloc((size_t)TT * DD * 2);
  _Float16* qh   = (_Float16*)alloc((size_t)TT * DD * 2);
  _Float16* kh   = (_Float16*)alloc((size_t)TT * DD * 2);
  _Float16* vth  = (_Float16*)alloc((size_t)TT * DD * 2);  // V^T [b][h][d][s]
  _Float16* ctxh = (_Float16*)alloc((size_t)TT * DD * 2);
  float*    hid  = (float*)   alloc((size_t)TT * DD * 4);
  _Float16* x2   = (_Float16*)alloc((size_t)TT * DD * 2);
  float*    lgt  = (float*)   alloc((size_t)TT * EE * 4);
  int*      e1   = (int*)     alloc((size_t)TT * 4);
  int*      e2   = (int*)     alloc((size_t)TT * 4);
  int*      s1   = (int*)     alloc((size_t)TT * 4);
  int*      s2   = (int*)     alloc((size_t)TT * 4);
  float*    g1   = (float*)   alloc((size_t)TT * 4);
  float*    g2   = (float*)   alloc((size_t)TT * 4);
  _Float16* xin  = (_Float16*)alloc((size_t)EE * CAP * DD * 2);
  _Float16* hm   = (_Float16*)alloc((size_t)EE * CAP * FF * 2);
  float*    eout = (float*)   alloc((size_t)EE * CAP * DD * 4);

  // 1) LN1
  ln_kernel<<<TT, 256, 0, stream>>>(hs, ln1w, ln1b, x1);

  // 2) Q/K/V projections (q scaled by 1/sqrt(64)); V stored transposed
  dim3 gqkv(DD / 64, TT / 256, 1);
  gemm_kernel<false, false><<<gqkv, 256, 0, stream>>>(
      x1, 0, DD, q_w, 0, DD, q_b, 0, 0.125f, qh, nullptr, 0, DD, nullptr, DD);
  gemm_kernel<false, false><<<gqkv, 256, 0, stream>>>(
      x1, 0, DD, k_w, 0, DD, k_b, 0, 1.0f, kh, nullptr, 0, DD, nullptr, DD);
  gemm_kernel<false, true><<<gqkv, 256, 0, stream>>>(
      x1, 0, DD, v_w, 0, DD, v_b, 0, 1.0f, vth, nullptr, 0, DD, nullptr, DD);

  // 3) causal flash attention
  attn_kernel<<<dim3(SS / 16, BB * HH), 32, 0, stream>>>(qh, kh, vth, ctxh);

  // 4) output projection + residual  -> hidden (f32)
  gemm_kernel<false, false><<<gqkv, 256, 0, stream>>>(
      ctxh, 0, DD, o_w, 0, DD, o_b, 0, 1.0f, nullptr, hid, 0, DD, hs, DD);

  // 5) LN2
  ln_kernel<<<TT, 256, 0, stream>>>(hid, ln2w, ln2b, x2);

  // 6) router logits + top-2 capacity routing
  router_kernel<<<TT / 16, 32, 0, stream>>>(x2, rw, lgt);
  route_kernel<<<1, 256, 0, stream>>>(lgt, e1, e2, s1, s2, g1, g2);

  // 7) dispatch tokens into expert slots
  zero_f16_kernel<<<4096, 256, 0, stream>>>(xin, (long)EE * CAP * DD);
  gather_kernel<<<TT, 256, 0, stream>>>(x2, e1, s1, e2, s2, xin);

  // 8) expert fc1 (+exact GELU): M=CAP=256 -> weights streamed exactly once
  gemm_kernel<true, false><<<dim3(FF / 64, CAP / 256, EE), 256, 0, stream>>>(
      xin, (long)CAP * DD, DD, fc1w, (long)DD * FF, FF, fc1b, FF, 1.0f,
      hm, nullptr, (long)CAP * FF, FF, nullptr, DD);

  // 9) expert fc2 (+bias, *0.8 eval-mode token dropout)
  gemm_kernel<false, false><<<dim3(DD / 64, CAP / 256, EE), 256, 0, stream>>>(
      hm, (long)CAP * FF, FF, fc2w, (long)FF * DD, DD, fc2b, DD, 0.8f,
      nullptr, eout, (long)CAP * DD, DD, nullptr, FF);

  // 10) combine + residual -> out
  combine_kernel<<<TT, 256, 0, stream>>>(hid, eout, e1, s1, e2, s2, g1, g2, out);
}
